// EdgellmAttention_10393820856971
// MI455X (gfx1250) — matrix operand order repaired and checked by reading, not compiled, not run on hardware
//
#include <hip/hip_runtime.h>
#include <math.h>

typedef __attribute__((ext_vector_type(16))) __bf16 v16bf;
typedef __attribute__((ext_vector_type(8)))  __bf16 v8bf;
typedef __attribute__((ext_vector_type(8)))  float  v8f;
typedef __attribute__((ext_vector_type(4)))  unsigned int v4u;
typedef __attribute__((ext_vector_type(8)))  int v8i;
typedef __attribute__((ext_vector_type(4)))  int v4i;

#define H_     16
#define D_NOPE 128
#define D_ROPE 64
#define D_QK   192
#define D_V    128
#define L_KV   512
#define HID_   2048
#define T_     4096

static constexpr float SCALING = 0.07216878364870322f;    // 1/sqrt(192)
static constexpr float LN1E4_OVER_HALF = 0.28782313662425572f; // ln(10000)/32

__device__ inline v16bf cat16(v8bf lo, v8bf hi) {
  return __builtin_shufflevector(lo, hi, 0,1,2,3,4,5,6,7,8,9,10,11,12,13,14,15);
}

// ---------------------------------------------------------------------------
// Tensor Data Mover: async 2-D bf16 tile load, global -> LDS, with per-row
// LDS padding. Descriptor per CDNA5 ISA §8.3-8.6. Issue from ONE wave; wait
// with s_wait_tensorcnt before the consuming barrier.
//   tile_x / tile_y      : tile extent in elements / rows
//   row_stride           : global row stride in elements
//   tensor_x / tensor_y  : tensor extents (OOB clamp; tiles here always fit)
//   pad_int / pad_amt    : ISA pad codes (interval 2^(c+1) DW, amount c+1 DW)
// ---------------------------------------------------------------------------
__device__ inline void tdm_load_2d(unsigned lds_off, const void* gptr,
                                   unsigned tile_x, unsigned tile_y,
                                   unsigned row_stride,
                                   unsigned tensor_x, unsigned tensor_y,
                                   unsigned pad_int, unsigned pad_amt) {
  unsigned long long ga = (unsigned long long)(size_t)gptr;
  v4u g0;
  g0[0] = 1u;                                          // count=1, user mode
  g0[1] = lds_off;                                     // LDS byte address
  g0[2] = (unsigned)(ga & 0xFFFFFFFFu);                // global addr lo
  g0[3] = (unsigned)((ga >> 32) & 0x1FFFFFFu) | (2u << 30); // addr hi | type=2
  v8i g1;
  g1[0] = (int)((1u << 16)                             // data_size = 2 bytes
              | (1u << 20)                             // pad_enable
              | (pad_int << 22) | (pad_amt << 25));
  g1[1] = (int)((tensor_x & 0xFFFFu) << 16);           // tensor_dim0 lo16
  g1[2] = (int)(((tensor_x >> 16) & 0xFFFFu) | ((tensor_y & 0xFFFFu) << 16));
  g1[3] = (int)(((tensor_y >> 16) & 0xFFFFu) | ((tile_x & 0xFFFFu) << 16));
  g1[4] = (int)(tile_y & 0xFFFFu);                     // tile_dim1 (dim2 = 0)
  g1[5] = (int)row_stride;                             // dim0_stride lo32
  g1[6] = 0;
  g1[7] = 0;
  v4i z4 = {};
  v8i z8 = {};
  __builtin_amdgcn_tensor_load_to_lds(g0, g1, z4, z4, z8, 0);
}

// ---------------------------------------------------------------------------
// f32 -> bf16 conversion
// ---------------------------------------------------------------------------
__global__ void cvt_f32_bf16_kernel(const float* __restrict__ s,
                                    __bf16* __restrict__ d, int n) {
  int i = blockIdx.x * blockDim.x + threadIdx.x;
  if (i < n) d[i] = (__bf16)s[i];
}

// ---------------------------------------------------------------------------
// bf16 WMMA GEMM: C[MxN] = A[MxK] @ B[KxN], row-major, f32 or bf16 out.
// 256x128 tile per 256-thread block; 8 waves in 4x2 grid; each wave a 64x64
// patch = 4x4 WMMA tiles. LDS double-buffered, 1 barrier per 32-K step.
// A tile staged by the Tensor Data Mover (wave 0), B staged manually
// (transposed scatter).
// ---------------------------------------------------------------------------
#define SA_SZ (256 * 40)
#define SB_SZ (128 * 40)

__global__ __launch_bounds__(256) void gemm_bf16_256x128(
    const __bf16* __restrict__ A, const __bf16* __restrict__ B,
    float* __restrict__ Cf, __bf16* __restrict__ Cb,
    int M, int N, int K)
{
  __shared__ __align__(16) __bf16 sA[2 * SA_SZ];   // [m][k], stride 40
  __shared__ __align__(16) __bf16 sB[2 * SB_SZ];   // B transposed: [n][k]

  const int tid  = threadIdx.x;
  const int wave = tid >> 5, lane = tid & 31;
  const int wr   = wave >> 1, wc = wave & 1;
  const int m0   = blockIdx.y * 256;
  const int n0   = blockIdx.x * 128;
  const int lr   = lane & 15, lh = lane >> 4;

  int bKk[2], bNoff[2];
  #pragma unroll
  for (int i = 0; i < 2; ++i) {
    int c = tid + i * 256;            // 512 chunks of 8 (32x128)
    bKk[i] = c >> 4; bNoff[i] = (c & 15) * 8;
  }

  v8f acc[4][4] = {};

  // ---- prologue: stage k0 = 0 into buffer 0 ----
  if (wave == 0)
    tdm_load_2d((unsigned)(size_t)sA, A + (size_t)m0 * K,
                /*tile_x=*/32, /*tile_y=*/256, /*row_stride=*/K,
                /*tensor_x=*/K, /*tensor_y=*/M,
                /*pad_int=*/3 /*16 DW*/, /*pad_amt=*/3 /*4 DW = 8 elems*/);
  #pragma unroll
  for (int i = 0; i < 2; ++i) {
    int col = n0 + bNoff[i];
    if (col < N) {
      v8bf v = *(const v8bf*)(B + (size_t)bKk[i] * N + col);
      #pragma unroll
      for (int j = 0; j < 8; ++j) sB[(bNoff[i] + j) * 40 + bKk[i]] = v[j];
    } else {
      #pragma unroll
      for (int j = 0; j < 8; ++j) sB[(bNoff[i] + j) * 40 + bKk[i]] = (__bf16)0.0f;
    }
  }
  if (wave == 0) __builtin_amdgcn_s_wait_tensorcnt(0);
  __syncthreads();

  int buf = 0;
  for (int k0 = 0; k0 < K; k0 += 32) {
    const __bf16* sAc = sA + buf * SA_SZ;
    const __bf16* sBc = sB + buf * SB_SZ;
    __bf16* sBn = sB + (buf ^ 1) * SB_SZ;
    const bool hasNext = (k0 + 32) < K;

    // kick off next A tile DMA + issue next B global loads early
    v8bf bReg[2];
    bool bOk[2] = {false, false};
    if (hasNext) {
      const int kn = k0 + 32;
      if (wave == 0)
        tdm_load_2d((unsigned)(size_t)(sA + (buf ^ 1) * SA_SZ),
                    A + (size_t)m0 * K + kn, 32, 256, K, K, M, 3, 3);
      #pragma unroll
      for (int i = 0; i < 2; ++i) {
        int col = n0 + bNoff[i];
        bOk[i] = col < N;
        if (bOk[i]) bReg[i] = *(const v8bf*)(B + (size_t)(kn + bKk[i]) * N + col);
      }
    }

    // compute: 4x4 tiles from current buffers
    v16bf bfr[4];
    #pragma unroll
    for (int c = 0; c < 4; ++c) {
      int n = wc * 64 + c * 16 + lr;
      int kb = lh * 16;
      bfr[c] = cat16(*(const v8bf*)(sBc + n * 40 + kb),
                     *(const v8bf*)(sBc + n * 40 + kb + 8));
    }
    #pragma unroll
    for (int r = 0; r < 4; ++r) {
      int m = wr * 64 + r * 16 + lr;
      int kb = lh * 16;
      v16bf af = cat16(*(const v8bf*)(sAc + m * 40 + kb),
                       *(const v8bf*)(sAc + m * 40 + kb + 8));
      #pragma unroll
      for (int c = 0; c < 4; ++c)
        acc[r][c] = __builtin_amdgcn_wmma_f32_16x16x32_bf16(
            false, af, false, bfr[c], (short)0, acc[r][c], false, false);
    }

    // commit next B tile to the other buffer
    if (hasNext) {
      #pragma unroll
      for (int i = 0; i < 2; ++i) {
        if (bOk[i]) {
          #pragma unroll
          for (int j = 0; j < 8; ++j) sBn[(bNoff[i] + j) * 40 + bKk[i]] = bReg[i][j];
        } else {
          #pragma unroll
          for (int j = 0; j < 8; ++j) sBn[(bNoff[i] + j) * 40 + bKk[i]] = (__bf16)0.0f;
        }
      }
      if (wave == 0) __builtin_amdgcn_s_wait_tensorcnt(0);
    }
    buf ^= 1;
    __syncthreads();
  }

  #pragma unroll
  for (int r = 0; r < 4; ++r)
    for (int c = 0; c < 4; ++c) {
      const int mt = m0 + wr * 64 + r * 16 + lh * 8;
      const int nt = n0 + wc * 64 + c * 16 + lr;
      if (nt < N) {
        if (Cf) {
          #pragma unroll
          for (int i = 0; i < 8; ++i) Cf[(size_t)(mt + i) * N + nt] = acc[r][c][i];
        } else {
          #pragma unroll
          for (int i = 0; i < 8; ++i) Cb[(size_t)(mt + i) * N + nt] = (__bf16)acc[r][c][i];
        }
      }
    }
}

// ---------------------------------------------------------------------------
// RMSNorm(latent[:,:512]) -> bf16 kv_a ; RoPE(latent[:,512:576]) -> bf16 k_pe
// ---------------------------------------------------------------------------
__global__ __launch_bounds__(256) void prep_latent_kernel(
    const float* __restrict__ lat, const float* __restrict__ lnw,
    const int* __restrict__ pos,
    __bf16* __restrict__ kva, __bf16* __restrict__ kpe)
{
  __shared__ float red[256];
  const int t = blockIdx.x, tid = threadIdx.x;
  const float* row = lat + (size_t)t * (L_KV + D_ROPE);
  float a = row[tid], b = row[tid + 256];
  red[tid] = a * a + b * b;
  __syncthreads();
  for (int s = 128; s > 0; s >>= 1) {
    if (tid < s) red[tid] += red[tid + s];
    __syncthreads();
  }
  float inv = rsqrtf(red[0] / (float)L_KV + 1e-6f);
  kva[(size_t)t * L_KV + tid]       = (__bf16)(a * inv * lnw[tid]);
  kva[(size_t)t * L_KV + tid + 256] = (__bf16)(b * inv * lnw[tid + 256]);
  if (tid < 32) {
    float p = (float)pos[t];
    float ang = p * __expf(-LN1E4_OVER_HALF * (float)tid);
    float sn, cs;
    __sincosf(ang, &sn, &cs);
    float x1 = row[L_KV + tid], x2 = row[L_KV + 32 + tid];
    kpe[(size_t)t * D_ROPE + tid]      = (__bf16)(x1 * cs - x2 * sn);
    kpe[(size_t)t * D_ROPE + 32 + tid] = (__bf16)(x2 * cs + x1 * sn);
  }
}

// ---------------------------------------------------------------------------
// q (T, H*192) f32 -> bf16 with RoPE applied to last 64 dims of each head
// ---------------------------------------------------------------------------
__global__ __launch_bounds__(256) void prep_q_kernel(
    const float* __restrict__ qf, const int* __restrict__ pos,
    __bf16* __restrict__ qb)
{
  const int t = blockIdx.x, tid = threadIdx.x;
  const float* row = qf + (size_t)t * (H_ * D_QK);
  __bf16* orow = qb + (size_t)t * (H_ * D_QK);
  #pragma unroll
  for (int i = 0; i < 8; ++i) {           // 16 heads * 128 nope = 2048 elems
    int idx = tid + i * 256;
    int h = idx >> 7, d = idx & 127;
    orow[h * D_QK + d] = (__bf16)row[h * D_QK + d];
  }
  float p = (float)pos[t];
  #pragma unroll
  for (int i = 0; i < 2; ++i) {           // 16 heads * 32 pairs = 512
    int pid = tid + i * 256;
    int h = pid >> 5, j = pid & 31;
    int base = h * D_QK + D_NOPE;
    float ang = p * __expf(-LN1E4_OVER_HALF * (float)j);
    float sn, cs;
    __sincosf(ang, &sn, &cs);
    float x1 = row[base + j], x2 = row[base + 32 + j];
    orow[base + j]      = (__bf16)(x1 * cs - x2 * sn);
    orow[base + 32 + j] = (__bf16)(x2 * cs + x1 * sn);
  }
}

// ---------------------------------------------------------------------------
// Flash attention: one block per (head, 128-query block).
// Q 128x192 resident in LDS (TDM-staged); loop over 64-key blocks (causal),
// online softmax. S = Q K^T and O += P V via bf16 WMMA. K tile TDM-staged.
// ---------------------------------------------------------------------------
__global__ __launch_bounds__(256) void flash_kernel(
    const __bf16* __restrict__ Q,    // (T, H*192)
    const __bf16* __restrict__ KV,   // (T, H*256) per head [k_nope|v]
    const __bf16* __restrict__ KPE,  // (T, 64)
    __bf16* __restrict__ O)          // (T, H*128)
{
  __shared__ __align__(16) __bf16 sQ[128 * 200];
  __shared__ __align__(16) __bf16 sK[64 * 200];
  __shared__ __align__(16) __bf16 sVt[128 * 72];  // [dv][key]
  __shared__ __align__(16) float  sS[128 * 65];
  __shared__ __align__(16) __bf16 sP[128 * 72];
  __shared__ float sM[128], sL[128], sScale[128];

  const int h = blockIdx.x, qb = blockIdx.y;
  const int tid = threadIdx.x, wave = tid >> 5, lane = tid & 31;
  const int wr = wave >> 1, wc = wave & 1;
  const int lr = lane & 15, lh = lane >> 4;
  const int t0 = qb * 128;

  // stage Q via TDM: cols 0..127 (pad 36 DW) + cols 128..191 (pad 68 DW)
  if (wave == 0) {
    tdm_load_2d((unsigned)(size_t)sQ,
                Q + (size_t)t0 * (H_ * D_QK) + h * D_QK,
                128, 128, H_ * D_QK, H_ * D_QK, T_, 5, 35);
    tdm_load_2d((unsigned)(size_t)(sQ + 128),
                Q + (size_t)t0 * (H_ * D_QK) + h * D_QK + 128,
                64, 128, H_ * D_QK, H_ * D_QK, T_, 4, 67);
  }
  if (tid < 128) { sM[tid] = -__builtin_inff(); sL[tid] = 0.0f; }
  if (wave == 0) __builtin_amdgcn_s_wait_tensorcnt(0);

  v8f o[2][4] = {};
  const int nkb = 2 * qb + 2;
  for (int kb = 0; kb < nkb; ++kb) {
    const int s0 = kb * 64;
    __syncthreads();
    // stage K tile via TDM: k_nope (64x128) + k_pe (64x64), LDS stride 200
    if (wave == 0) {
      tdm_load_2d((unsigned)(size_t)sK,
                  KV + (size_t)s0 * (H_ * 2 * D_NOPE) + h * (2 * D_NOPE),
                  128, 64, H_ * 2 * D_NOPE, H_ * 2 * D_NOPE, T_, 5, 35);
      tdm_load_2d((unsigned)(size_t)(sK + 128),
                  KPE + (size_t)s0 * D_ROPE,
                  64, 64, D_ROPE, D_ROPE, T_, 4, 67);
    }
    // stage V transposed: sVt[dv][key]
    #pragma unroll
    for (int i = 0; i < 32; ++i) {
      int idx = tid + i * 256;            // 8192 scalars
      int r = idx >> 7, dv = idx & 127;
      sVt[dv * 72 + r] = KV[(size_t)(s0 + r) * (H_ * 2 * D_NOPE)
                            + h * (2 * D_NOPE) + D_NOPE + dv];
    }
    if (wave == 0) __builtin_amdgcn_s_wait_tensorcnt(0);
    __syncthreads();

    // S = Q K^T  (128q x 64k): each wave 2x2 of 16x16 tiles
    v8f sacc[2][2] = {};
    #pragma unroll
    for (int ds = 0; ds < 6; ++ds) {
      v16bf af[2], bk[2];
      #pragma unroll
      for (int r = 0; r < 2; ++r) {
        int m = wr * 32 + r * 16 + lr;
        int kbo = ds * 32 + lh * 16;
        af[r] = cat16(*(const v8bf*)(sQ + m * 200 + kbo),
                      *(const v8bf*)(sQ + m * 200 + kbo + 8));
      }
      #pragma unroll
      for (int c = 0; c < 2; ++c) {
        int n = wc * 32 + c * 16 + lr;
        int kbo = ds * 32 + lh * 16;
        bk[c] = cat16(*(const v8bf*)(sK + n * 200 + kbo),
                      *(const v8bf*)(sK + n * 200 + kbo + 8));
      }
      #pragma unroll
      for (int r = 0; r < 2; ++r)
        #pragma unroll
        for (int c = 0; c < 2; ++c)
          sacc[r][c] = __builtin_amdgcn_wmma_f32_16x16x32_bf16(
              false, af[r], false, bk[c], (short)0, sacc[r][c], false, false);
    }
    // spill scaled S to LDS
    #pragma unroll
    for (int r = 0; r < 2; ++r)
      #pragma unroll
      for (int c = 0; c < 2; ++c) {
        int rowb = wr * 32 + r * 16 + lh * 8;
        int col  = wc * 32 + c * 16 + lr;
        #pragma unroll
        for (int i = 0; i < 8; ++i)
          sS[(rowb + i) * 65 + col] = sacc[r][c][i] * SCALING;
      }
    __syncthreads();

    // online softmax row pass (one thread per query row)
    if (tid < 128) {
      int r = tid, tg = t0 + r;
      float mold = sM[r], mnew = mold;
      for (int c = 0; c < 64; ++c) {
        float v = (s0 + c <= tg) ? sS[r * 65 + c] : -__builtin_inff();
        mnew = fmaxf(mnew, v);
      }
      float corr = __expf(mold - mnew);
      float l = sL[r] * corr;
      for (int c = 0; c < 64; ++c) {
        float p = (s0 + c <= tg) ? __expf(sS[r * 65 + c] - mnew) : 0.0f;
        sP[r * 72 + c] = (__bf16)p;
        l += p;
      }
      sM[r] = mnew; sL[r] = l; sScale[r] = corr;
    }
    __syncthreads();

    // rescale O accumulators by per-row correction
    #pragma unroll
    for (int r = 0; r < 2; ++r) {
      int rowb = wr * 32 + r * 16 + lh * 8;
      #pragma unroll
      for (int i = 0; i < 8; ++i) {
        float f = sScale[rowb + i];
        #pragma unroll
        for (int c = 0; c < 4; ++c) o[r][c][i] *= f;
      }
    }
    // O += P V   (128q x 128dv, K=64 keys): each wave 2x4 tiles
    #pragma unroll
    for (int ks = 0; ks < 2; ++ks) {
      v16bf af[2], bv[4];
      #pragma unroll
      for (int r = 0; r < 2; ++r) {
        int m = wr * 32 + r * 16 + lr;
        int kbo = ks * 32 + lh * 16;
        af[r] = cat16(*(const v8bf*)(sP + m * 72 + kbo),
                      *(const v8bf*)(sP + m * 72 + kbo + 8));
      }
      #pragma unroll
      for (int c = 0; c < 4; ++c) {
        int n = wc * 64 + c * 16 + lr;
        int kbo = ks * 32 + lh * 16;
        bv[c] = cat16(*(const v8bf*)(sVt + n * 72 + kbo),
                      *(const v8bf*)(sVt + n * 72 + kbo + 8));
      }
      #pragma unroll
      for (int r = 0; r < 2; ++r)
        #pragma unroll
        for (int c = 0; c < 4; ++c)
          o[r][c] = __builtin_amdgcn_wmma_f32_16x16x32_bf16(
              false, af[r], false, bv[c], (short)0, o[r][c], false, false);
    }
  }
  __syncthreads();

  // epilogue: divide by l, store bf16 (T, H*128)
  #pragma unroll
  for (int r = 0; r < 2; ++r) {
    int rowb = wr * 32 + r * 16 + lh * 8;
    #pragma unroll
    for (int c = 0; c < 4; ++c) {
      int col = wc * 64 + c * 16 + lr;
      #pragma unroll
      for (int i = 0; i < 8; ++i) {
        float inv = 1.0f / sL[rowb + i];
        O[(size_t)(t0 + rowb + i) * (H_ * D_V) + h * D_V + col] =
            (__bf16)(o[r][c][i] * inv);
      }
    }
  }
}

// ---------------------------------------------------------------------------
extern "C" void kernel_launch(void* const* d_in, const int* in_sizes, int n_in,
                              void* d_out, int out_size, void* d_ws, size_t ws_size,
                              hipStream_t stream)
{
  const int*   pos    = (const int*)d_in[0];
  const float* hidden = (const float*)d_in[1];
  const float* Wq     = (const float*)d_in[2];
  const float* Wkva   = (const float*)d_in[3];
  const float* lnw    = (const float*)d_in[4];
  const float* Wkvb   = (const float*)d_in[5];
  const float* Wo     = (const float*)d_in[6];
  float* out = (float*)d_out;

  char* ws = (char*)d_ws;
  size_t off = 0;
  auto alloc = [&](size_t bytes) -> char* {
    char* p = ws + off;
    off += (bytes + 255) & ~(size_t)255;
    return p;
  };
  __bf16* hidden_bf = (__bf16*)alloc((size_t)T_ * HID_ * 2);
  __bf16* Wq_bf     = (__bf16*)alloc((size_t)HID_ * (H_ * D_QK) * 2);
  __bf16* Wkva_bf   = (__bf16*)alloc((size_t)HID_ * (L_KV + D_ROPE) * 2);
  __bf16* Wkvb_bf   = (__bf16*)alloc((size_t)L_KV * (H_ * 2 * D_NOPE) * 2);
  __bf16* Wo_bf     = (__bf16*)alloc((size_t)(H_ * D_V) * HID_ * 2);
  float*  qf        = (float*)alloc((size_t)T_ * (H_ * D_QK) * 4);
  float*  latf      = (float*)alloc((size_t)T_ * (L_KV + D_ROPE) * 4);
  __bf16* qbf       = (__bf16*)alloc((size_t)T_ * (H_ * D_QK) * 2);
  __bf16* kvabf     = (__bf16*)alloc((size_t)T_ * L_KV * 2);
  __bf16* kpebf     = (__bf16*)alloc((size_t)T_ * D_ROPE * 2);
  __bf16* kvbf      = (__bf16*)alloc((size_t)T_ * (H_ * 2 * D_NOPE) * 2);
  __bf16* attnbf    = (__bf16*)alloc((size_t)T_ * (H_ * D_V) * 2);

  auto cvt = [&](const float* s, __bf16* d, int n) {
    cvt_f32_bf16_kernel<<<(n + 255) / 256, 256, 0, stream>>>(s, d, n);
  };
  cvt(hidden, hidden_bf, T_ * HID_);
  cvt(Wq,     Wq_bf,     HID_ * (H_ * D_QK));
  cvt(Wkva,   Wkva_bf,   HID_ * (L_KV + D_ROPE));
  cvt(Wkvb,   Wkvb_bf,   L_KV * (H_ * 2 * D_NOPE));
  cvt(Wo,     Wo_bf,     (H_ * D_V) * HID_);

  auto gemm = [&](const __bf16* A, const __bf16* B, float* Cf, __bf16* Cb,
                  int M, int N, int K) {
    gemm_bf16_256x128<<<dim3((N + 127) / 128, M / 256), 256, 0, stream>>>(
        A, B, Cf, Cb, M, N, K);
  };

  // q = X @ Wq
  gemm(hidden_bf, Wq_bf, qf, nullptr, T_, H_ * D_QK, HID_);
  // latent = X @ Wkva
  gemm(hidden_bf, Wkva_bf, latf, nullptr, T_, L_KV + D_ROPE, HID_);
  // rmsnorm + rope(k_pe)
  prep_latent_kernel<<<T_, 256, 0, stream>>>(latf, lnw, pos, kvabf, kpebf);
  // kv = kv_a @ Wkvb  (bf16 out)
  gemm(kvabf, Wkvb_bf, nullptr, kvbf, T_, H_ * 2 * D_NOPE, L_KV);
  // rope(q) + pack bf16
  prep_q_kernel<<<T_, 256, 0, stream>>>(qf, pos, qbf);
  // flash attention
  flash_kernel<<<dim3(H_, T_ / 128), 256, 0, stream>>>(qbf, kvbf, kpebf, attnbf);
  // out = attn @ Wo  (f32 out)
  gemm(attnbf, Wo_bf, out, nullptr, T_, HID_, H_ * D_V);
}
